// SelfAttention_81956565942824
// MI455X (gfx1250) — compile-verified, matrix operand order checked by reading, not run on hardware
//
#include <hip/hip_runtime.h>

typedef __attribute__((ext_vector_type(16))) __bf16 v16bf;
typedef __attribute__((ext_vector_type(8)))  float  v8f;
typedef unsigned short u16;
typedef unsigned int   u32;
typedef int v4i __attribute__((vector_size(16)));   // async builtin param pointee

#define EMBED 1024
#define NHEAD 16
#define HDIM  64
#define SEQ   2048
#define BATCH 2
#define MROWS (BATCH*SEQ)      // 4096
#define FQKV  (3*EMBED)        // 3072

// ---- async LDS path (gfx1250 GLOBAL_LOAD_ASYNC_TO_LDS_*, ASYNCcnt) -------
#if defined(__has_builtin)
#if __has_builtin(__builtin_amdgcn_global_load_async_to_lds_b128) && \
    __has_builtin(__builtin_amdgcn_s_wait_asynccnt)
#define HAS_ASYNC_LDS 1
#endif
#if __has_builtin(__builtin_amdgcn_tensor_load_to_lds) && \
    __has_builtin(__builtin_amdgcn_s_wait_tensorcnt)
#define HAS_TDM 1
#endif
#endif

#ifdef HAS_ASYNC_LDS
#define CP16(dst, src)                                                        \
    __builtin_amdgcn_global_load_async_to_lds_b128(                           \
        (v4i*)(src), (v4i*)(dst), 0, 0)
#define WAIT_ASYNC(n) __builtin_amdgcn_s_wait_asynccnt(n)
#else
#define CP16(dst, src)                                                        \
    (*reinterpret_cast<uint4*>(dst) = *reinterpret_cast<const uint4*>(src))
#define WAIT_ASYNC(n) ((void)0)
#endif

#ifdef HAS_TDM
typedef __attribute__((ext_vector_type(4))) unsigned int v4u_t;
typedef __attribute__((ext_vector_type(8))) int         v8i_t;
typedef __attribute__((ext_vector_type(4))) int         v4i_t;

// TDM load of a 32(kv) x 64(d) bf16 tile from a row-major [2048,64] tensor into LDS.
// D# per CDNA5 ISA (2-D: groups 2/3 disabled/zero).
// This toolchain's builtin is the 6-arg form:
//   (uint32x4 g0, int32x8 g1, int32x4 g2, int32x4 g3, int32x8 pad, i32 cpol)
__device__ __forceinline__ void tdm_load_k(const u16* gsrc, u16* lds_dst) {
    const u32 lds_addr = (u32)(unsigned long long)(uintptr_t)lds_dst; // LDS byte offset
    const unsigned long long ga = (unsigned long long)(uintptr_t)gsrc;
    v4u_t g0;
    g0.x = 1u;                                   // count=1, user descriptor
    g0.y = lds_addr;                             // lds_addr[31:0]
    g0.z = (u32)(ga & 0xFFFFFFFFu);              // global_addr[31:0]
    g0.w = (u32)((ga >> 32) & 0x01FFFFFFu)       // global_addr[56:32]
         | 0x80000000u;                          // type=2 ("image") in bits 127:126
    v8i_t g1;
    g1[0] = 0x00010000;                          // data_size=1 (2 bytes/elem)
    g1[1] = (int)(64u   << 16);                  // tensor_dim0 = 64   (bits 79:48 lo)
    g1[2] = (int)(2048u << 16);                  // tensor_dim1 = 2048 (bits 111:80 lo)
    g1[3] = (int)(64u   << 16);                  // tile_dim0 = 64     (bits 127:112)
    g1[4] = 32;                                  // tile_dim1 = 32, tile_dim2 = 0
    g1[5] = 64;                                  // tensor_dim0_stride = 64 (lo 32)
    g1[6] = 0;                                   // stride hi / dim1_stride lo
    g1[7] = 0;
    v4i_t z4 = {0, 0, 0, 0};
    v8i_t z8 = {0, 0, 0, 0, 0, 0, 0, 0};
    __builtin_amdgcn_tensor_load_to_lds(g0, g1, z4, z4, z8, 0);
}
#define WAIT_TENSOR(n) __builtin_amdgcn_s_wait_tensorcnt(n)
#endif

// ---------- helpers ----------

__device__ __forceinline__ u16 f2bf(float f) {
    union { float f; u32 u; } x; x.f = f;
    u32 r = x.u + 0x7FFFu + ((x.u >> 16) & 1u);   // round-to-nearest-even
    return (u16)(r >> 16);
}

// Assemble a 16x32 bf16 A/B fragment per the CDNA5 16-bit matrix VGPR layout:
// lane half h: element pair j maps to K dword (j<4 ? j : j+4) + 4h.
__device__ __forceinline__ v16bf load_frag(const u16* base, int h) {
    union { v16bf v; u32 u[8]; } r;
    const u32* b32 = reinterpret_cast<const u32*>(base);
#pragma unroll
    for (int j = 0; j < 8; ++j) {
        int dw = (j < 4 ? j : j + 4) + 4 * h;
        r.u[j] = b32[dw];
    }
    return r.v;
}

__device__ __forceinline__ v8f wmma_bf16(v16bf a, v16bf b, v8f c) {
    return __builtin_amdgcn_wmma_f32_16x16x32_bf16(false, a, false, b, (short)0, c, false, false);
}

// ---------- kernel 1: fp32 -> bf16 pack ----------

__global__ __launch_bounds__(256) void cvt_bf16_kernel(const float* __restrict__ s,
                                                       u16* __restrict__ d, int n2) {
    int i = blockIdx.x * 256 + threadIdx.x;
    int stride = gridDim.x * 256;
    for (int j = i; j < n2; j += stride) {
        float2 f = reinterpret_cast<const float2*>(s)[j];
        u32 v = (u32)f2bf(f.x) | ((u32)f2bf(f.y) << 16);
        reinterpret_cast<u32*>(d)[j] = v;
    }
}

// ---------- kernel 2: QKV GEMM  out = X * Wqkv^T, scatter to Q,K and V^T ----------

__global__ __launch_bounds__(256) void qkv_gemm_kernel(const u16* __restrict__ xb,
                                                       const u16* __restrict__ wb,
                                                       u16* __restrict__ qo,
                                                       u16* __restrict__ ko,
                                                       u16* __restrict__ vto) {
    const int wid  = blockIdx.x * 8 + (threadIdx.x >> 5);
    const int lane = threadIdx.x & 31;
    const int lh   = lane >> 4;
    const int ln   = lane & 15;
    const int mt   = wid / (FQKV / 64);
    const int ng   = wid % (FQKV / 64);
    const int m0   = mt * 32;
    const int f0   = ng * 64;

    v8f acc[2][4] = {};
    const u16* arow0 = xb + (size_t)(m0 + ln) * EMBED;
    const u16* arow1 = xb + (size_t)(m0 + 16 + ln) * EMBED;
    const u16* brow  = wb + (size_t)(f0 + ln) * EMBED;

    for (int k0 = 0; k0 < EMBED; k0 += 32) {
        __builtin_prefetch(arow0 + k0 + 64, 0, 3);
        v16bf a0 = load_frag(arow0 + k0, lh);
        v16bf a1 = load_frag(arow1 + k0, lh);
#pragma unroll
        for (int t = 0; t < 4; ++t) {
            v16bf b = load_frag(brow + (size_t)t * 16 * EMBED + k0, lh);
            acc[0][t] = wmma_bf16(a0, b, acc[0][t]);
            acc[1][t] = wmma_bf16(a1, b, acc[1][t]);
        }
    }

    const int sec = f0 >> 10;            // 0=Q 1=K 2=V (uniform per wave)
    const int fin = f0 & 1023;
    const int hh  = fin >> 6;            // head (uniform)
    const int db  = fin & 63;            // d base within head

#pragma unroll
    for (int half = 0; half < 2; ++half) {
#pragma unroll
        for (int t = 0; t < 4; ++t) {
            const int d = db + t * 16 + ln;
#pragma unroll
            for (int r = 0; r < 8; ++r) {
                const int row = m0 + half * 16 + r + 8 * lh;   // token index
                const int bb  = row >> 11;
                const int nn  = row & 2047;
                const u16 val = f2bf(acc[half][t][r]);
                if (sec == 0)
                    qo[(((size_t)(bb * NHEAD + hh) * SEQ) + nn) * HDIM + d] = val;
                else if (sec == 1)
                    ko[(((size_t)(bb * NHEAD + hh) * SEQ) + nn) * HDIM + d] = val;
                else
                    vto[(((size_t)(bb * NHEAD + hh) * HDIM) + d) * SEQ + nn] = val;
            }
        }
    }
}

// ---------- kernel 3: flash attention ----------
// Block = 256 threads (8 waves); 128 query rows of one (b,h) per block.
// K tile staged by the Tensor Data Mover (TENSORcnt), Vᵀ tile by async-to-LDS
// loads (ASYNCcnt), both double-buffered so tile i+1 streams in while tile i
// is consumed by the WMMAs.

__global__ __launch_bounds__(256) void flash_attn_kernel(const u16* __restrict__ q,
                                                         const u16* __restrict__ k,
                                                         const u16* __restrict__ vt,
                                                         u16* __restrict__ aout) {
    const int bq = blockIdx.x;            // 0..15  (query block of 128)
    const int hh = blockIdx.y;            // head
    const int bb = blockIdx.z;            // batch
    const int wv   = threadIdx.x >> 5;
    const int lane = threadIdx.x & 31;
    const int lh   = lane >> 4;
    const int ln   = lane & 15;
    const int q0   = bq * 128 + wv * 16;

    const u16* Qb = q  + (size_t)(bb * NHEAD + hh) * SEQ * HDIM;
    const u16* Kb = k  + (size_t)(bb * NHEAD + hh) * SEQ * HDIM;
    const u16* Vb = vt + (size_t)(bb * NHEAD + hh) * HDIM * SEQ;

    // Q as B-operand of Sᵀ: lane = q column, elements = d (two halves of D=64)
    const v16bf bq0 = load_frag(Qb + (size_t)(q0 + ln) * HDIM + 0,  lh);
    const v16bf bq1 = load_frag(Qb + (size_t)(q0 + ln) * HDIM + 32, lh);

    float m_l = -1e30f, l_l = 0.0f;       // per-lane softmax state for q = ln
    v8f o[4] = {};

    __shared__ u16 sK[2][32 * HDIM];      // 32 kv rows x 64 d   (4 KB each)
    __shared__ u16 sV[2][HDIM * 32];      // 64 d rows x 32 kv   (Vᵀ tile)

    const int tid = threadIdx.x;
    // ---- stage tile 0 into buffer 0 ----
#ifdef HAS_TDM
    if (tid < 32) tdm_load_k(Kb, &sK[0][0]);
#else
    CP16(&sK[0][tid * 8], Kb + tid * 8);
#endif
    {
        int r = tid >> 2, c = tid & 3;
        CP16(&sV[0][r * 32 + c * 8], Vb + (size_t)r * SEQ + c * 8);
    }

    for (int kv0 = 0; kv0 < SEQ; kv0 += 32) {
        const int p   = (kv0 >> 5) & 1;
        const int kvn = (kv0 + 32 < SEQ) ? kv0 + 32 : kv0;   // last iter: dummy refill

        __syncthreads();                  // prior reads of buffer p^1 complete
        // ---- issue stage of next tile into the other buffer ----
#ifdef HAS_TDM
        if (tid < 32) tdm_load_k(Kb + (size_t)kvn * HDIM, &sK[p ^ 1][0]);
        {
            int r = tid >> 2, c = tid & 3;
            CP16(&sV[p ^ 1][r * 32 + c * 8], Vb + (size_t)r * SEQ + kvn + c * 8);
        }
        WAIT_ASYNC(1);                    // V tile i landed (only newest outstanding)
        if (tid < 32) WAIT_TENSOR(1);     // K tile i landed
#else
        CP16(&sK[p ^ 1][tid * 8], Kb + (size_t)kvn * HDIM + tid * 8);
        {
            int r = tid >> 2, c = tid & 3;
            CP16(&sV[p ^ 1][r * 32 + c * 8], Vb + (size_t)r * SEQ + kvn + c * 8);
        }
        WAIT_ASYNC(2);                    // only the 2 just-issued remain
#endif
        __syncthreads();                  // ... for every wave in the block

        // Sᵀ tiles: tile0 = kv rows [0,16), tile1 = [16,32); contraction over d.
        v16bf ak0a = load_frag(&sK[p][(0  + ln) * HDIM + 0],  lh);
        v16bf ak0b = load_frag(&sK[p][(0  + ln) * HDIM + 32], lh);
        v16bf ak1a = load_frag(&sK[p][(16 + ln) * HDIM + 0],  lh);
        v16bf ak1b = load_frag(&sK[p][(16 + ln) * HDIM + 32], lh);
        v8f s0 = {}; s0 = wmma_bf16(ak0a, bq0, s0); s0 = wmma_bf16(ak0b, bq1, s0);
        v8f s1 = {}; s1 = wmma_bf16(ak1a, bq0, s1); s1 = wmma_bf16(ak1b, bq1, s1);

        // scale + running row max (row of S == lane-local values + other half)
        const float scale = 0.125f;       // 1/sqrt(64)
        float mx = m_l;
#pragma unroll
        for (int r = 0; r < 8; ++r) {
            s0[r] *= scale; s1[r] *= scale;
            mx = fmaxf(mx, fmaxf(s0[r], s1[r]));
        }
        mx = fmaxf(mx, __shfl_xor(mx, 16, 32));
        const float corr = __expf(m_l - mx);

        // exp + pack straight into the P·V A-fragment
        union { v16bf v; u16 s[16]; } ap;
        float sum = 0.0f;
#pragma unroll
        for (int r = 0; r < 8; ++r) {
            float p0 = __expf(s0[r] - mx);
            float p1 = __expf(s1[r] - mx);
            sum += p0 + p1;
            ap.s[r]     = f2bf(p0);
            ap.s[8 + r] = f2bf(p1);
        }
        sum += __shfl_xor(sum, 16, 32);
        l_l = l_l * corr + sum;
        m_l = mx;

        // rescale O (O rows live at q = r + 8*lh; corr for that q sits in lane r+8*lh)
#pragma unroll
        for (int r = 0; r < 8; ++r) {
            float c = __shfl(corr, r + 8 * lh, 32);
#pragma unroll
            for (int t = 0; t < 4; ++t) o[t][r] *= c;
        }

        // P·V: B operand from Vᵀ tile (lane = d, elements = kv pairs)
#pragma unroll
        for (int t = 0; t < 4; ++t) {
            v16bf bv = load_frag(&sV[p][(t * 16 + ln) * 32], lh);
            o[t] = wmma_bf16(ap.v, bv, o[t]);
        }
    }

    // normalize + store concat-head bf16 [B*N, E]
    const float inv = 1.0f / l_l;
#pragma unroll
    for (int r = 0; r < 8; ++r) {
        float iv = __shfl(inv, r + 8 * lh, 32);
        const size_t row = (size_t)bb * SEQ + q0 + r + 8 * lh;
#pragma unroll
        for (int t = 0; t < 4; ++t)
            aout[row * EMBED + hh * HDIM + t * 16 + ln] = f2bf(o[t][r] * iv);
    }
}

// ---------- kernel 4: output projection  out = A * Wproj^T + b ----------

__global__ __launch_bounds__(256) void proj_gemm_kernel(const u16* __restrict__ ab,
                                                        const u16* __restrict__ wb,
                                                        const float* __restrict__ bias,
                                                        float* __restrict__ out) {
    const int wid  = blockIdx.x * 8 + (threadIdx.x >> 5);
    const int lane = threadIdx.x & 31;
    const int lh   = lane >> 4;
    const int ln   = lane & 15;
    const int mt   = wid / (EMBED / 64);
    const int ng   = wid % (EMBED / 64);
    const int m0   = mt * 32;
    const int f0   = ng * 64;

    v8f acc[2][4] = {};
    const u16* arow0 = ab + (size_t)(m0 + ln) * EMBED;
    const u16* arow1 = ab + (size_t)(m0 + 16 + ln) * EMBED;
    const u16* brow  = wb + (size_t)(f0 + ln) * EMBED;

    for (int k0 = 0; k0 < EMBED; k0 += 32) {
        __builtin_prefetch(arow0 + k0 + 64, 0, 3);
        v16bf a0 = load_frag(arow0 + k0, lh);
        v16bf a1 = load_frag(arow1 + k0, lh);
#pragma unroll
        for (int t = 0; t < 4; ++t) {
            v16bf b = load_frag(brow + (size_t)t * 16 * EMBED + k0, lh);
            acc[0][t] = wmma_bf16(a0, b, acc[0][t]);
            acc[1][t] = wmma_bf16(a1, b, acc[1][t]);
        }
    }

#pragma unroll
    for (int half = 0; half < 2; ++half) {
#pragma unroll
        for (int t = 0; t < 4; ++t) {
            const int f  = f0 + t * 16 + ln;
            const float bv = bias[f];
#pragma unroll
            for (int r = 0; r < 8; ++r) {
                const size_t row = (size_t)m0 + half * 16 + r + 8 * lh;
                out[row * EMBED + f] = acc[half][t][r] + bv;
            }
        }
    }
}

// ---------- launch ----------

extern "C" void kernel_launch(void* const* d_in, const int* in_sizes, int n_in,
                              void* d_out, int out_size, void* d_ws, size_t ws_size,
                              hipStream_t stream) {
    const float* x      = (const float*)d_in[0];   // [2,2048,1024]
    const float* w_qkv  = (const float*)d_in[1];   // [3072,1024]
    const float* w_proj = (const float*)d_in[2];   // [1024,1024]
    const float* b_proj = (const float*)d_in[3];   // [1024]
    float* out = (float*)d_out;

    char* ws = (char*)d_ws;
    u16* xb    = (u16*)(ws + 0);                       //  8 MB  [4096,1024]
    u16* wqkvb = (u16*)(ws + (8u  << 20));             //  6 MB  [3072,1024]
    u16* wprjb = (u16*)(ws + (14u << 20));             //  2 MB  [1024,1024]
    u16* qb    = (u16*)(ws + (16u << 20));             //  8 MB  [B,H,N,D]
    u16* kb    = (u16*)(ws + (24u << 20));             //  8 MB  [B,H,N,D]
    u16* vtb   = (u16*)(ws + (32u << 20));             //  8 MB  [B,H,D,N]
    u16* attnb = (u16*)(ws + (40u << 20));             //  8 MB  [4096,1024]

    cvt_bf16_kernel<<<1024, 256, 0, stream>>>(x,      xb,    MROWS * EMBED / 2);
    cvt_bf16_kernel<<<1024, 256, 0, stream>>>(w_qkv,  wqkvb, FQKV  * EMBED / 2);
    cvt_bf16_kernel<<<1024, 256, 0, stream>>>(w_proj, wprjb, EMBED * EMBED / 2);

    // (4096/32) * (3072/64) = 6144 waves, 8 waves/block
    qkv_gemm_kernel<<<768, 256, 0, stream>>>(xb, wqkvb, qb, kb, vtb);

    flash_attn_kernel<<<dim3(SEQ / 128, NHEAD, BATCH), 256, 0, stream>>>(qb, kb, vtb, attnb);

    // (4096/32) * (1024/64) = 2048 waves, 8 waves/block
    proj_gemm_kernel<<<256, 256, 0, stream>>>(attnb, wprjb, b_proj, out);
}
